// MultiHeadedAttention_38903813767258
// MI455X (gfx1250) — matrix operand order repaired and checked by reading, not compiled, not run on hardware
//
#include <hip/hip_runtime.h>

// ---------------- WMMA fragment types & helpers (gfx1250, wave32) ----------------
typedef __attribute__((ext_vector_type(16))) __bf16 v16bf;
typedef __attribute__((ext_vector_type(8)))  float  v8f;
typedef __attribute__((ext_vector_type(4)))  float  fvec4;

// A-matrix 16x32 bf16 layout (same k order as B):
//   lanes 0-15 : M = lane, VGPR j holds K = 2j, 2j+1
//   lanes 16-31: M = lane-16, VGPR j holds K = 16+2j, 16+2j+1
__device__ __forceinline__ int koffA(int j, int half) {
  return 2 * j + 16 * half;
}
// B-matrix 32x16 bf16 layout (contiguous k order, same as A):
//   lanes 0-15 : N = lane, VGPR j holds K = 2j, 2j+1
//   lanes 16-31: N = lane-16, VGPR j holds K = 16+2j, 16+2j+1
__device__ __forceinline__ int koffB(int j, int half) {
  return 2 * j + 16 * half;
}

// Load A fragment from row-major [M x K] source (rows = m, stride ld).
template <typename T>
__device__ __forceinline__ v16bf load_a(const T* base, int ld, int lane) {
  const int m = lane & 15, half = lane >> 4;
  const T* row = base + (size_t)m * ld;
  v16bf f;
#pragma unroll
  for (int j = 0; j < 8; ++j) {
    const int k = koffA(j, half);
    f[2 * j]     = (__bf16)row[k];
    f[2 * j + 1] = (__bf16)row[k + 1];
  }
  return f;
}

// Load B fragment where logical B[k][n] = src[n*ld + k]  (src row-major [N x K],
// i.e. a transposed operand: weight matrices stored [out,in], or K for Q*K^T).
template <typename T>
__device__ __forceinline__ v16bf load_bt(const T* base, int ld, int lane) {
  const int n = lane & 15, half = lane >> 4;
  const T* row = base + (size_t)n * ld;
  v16bf f;
#pragma unroll
  for (int j = 0; j < 8; ++j) {
    const int k = koffB(j, half);
    f[2 * j]     = (__bf16)row[k];
    f[2 * j + 1] = (__bf16)row[k + 1];
  }
  return f;
}

// Load B fragment where logical B[k][n] = src[k*ld + n]  (natural row-major K x N, e.g. V).
template <typename T>
__device__ __forceinline__ v16bf load_b(const T* base, int ld, int lane) {
  const int n = lane & 15, half = lane >> 4;
  v16bf f;
#pragma unroll
  for (int j = 0; j < 8; ++j) {
    const int k = koffB(j, half);
    f[2 * j]     = (__bf16)base[(size_t)k * ld + n];
    f[2 * j + 1] = (__bf16)base[(size_t)(k + 1) * ld + n];
  }
  return f;
}

__device__ __forceinline__ v8f wmma_bf16(v16bf a, v16bf b, v8f c) {
  return __builtin_amdgcn_wmma_f32_16x16x32_bf16(false, a, false, b, (short)0, c,
                                                 false, false);
}

// ---------------- constants ----------------
#define BB   2
#define SS   2048
#define HID  768
#define HH   12
#define DD   64

// ---------------- kernel 1: QKV projection -> bf16 [B,H,S,D] ----------------
// grid = (M/16, HID/64, 3), block = 128 (4 waves). y = x @ W^T + b
__global__ __launch_bounds__(128) void qkv_proj_kernel(
    const float* __restrict__ xq, const float* __restrict__ xk,
    const float* __restrict__ xv,
    const float* __restrict__ Wq, const float* __restrict__ bq,
    const float* __restrict__ Wk, const float* __restrict__ bk,
    const float* __restrict__ Wv, const float* __restrict__ bv,
    __bf16* __restrict__ qo, __bf16* __restrict__ ko, __bf16* __restrict__ vo) {
  const int lane = threadIdx.x & 31, wave = threadIdx.x >> 5;
  const int m0 = blockIdx.x * 16;
  const int n0 = blockIdx.y * 64 + wave * 16;

  const float* x;
  const float* W;
  const float* bias;
  __bf16* out;
  if (blockIdx.z == 0)      { x = xq; W = Wq; bias = bq; out = qo; }
  else if (blockIdx.z == 1) { x = xk; W = Wk; bias = bk; out = ko; }
  else                      { x = xv; W = Wv; bias = bv; out = vo; }

  v8f c = {};
  for (int kk = 0; kk < HID; kk += 32) {
    v16bf a = load_a(x + (size_t)m0 * HID + kk, HID, lane);
    v16bf b = load_bt(W + (size_t)n0 * HID + kk, HID, lane);
    c = wmma_bf16(a, b, c);
  }

  const int n = lane & 15, half = lane >> 4;
  const float bn = bias[n0 + n];
#pragma unroll
  for (int r = 0; r < 8; ++r) {
    const int m = m0 + r + 8 * half;   // global row in [B*S]
    const int ng = n0 + n;             // global col in [H*D]
    const int bidx = m >> 11, s = m & (SS - 1);
    const int h = ng >> 6, d = ng & (DD - 1);
    out[(((size_t)bidx * HH + h) * SS + s) * DD + d] = (__bf16)(c[r] + bn);
  }
}

// ---------------- kernel 2: fused attention core ----------------
// grid = (S/16, H, B), block = 128 (4 waves), dyn LDS = 16*2048*4 + 16*2048*2
__global__ __launch_bounds__(128) void attn_core_kernel(
    const __bf16* __restrict__ qws, const __bf16* __restrict__ kws,
    const __bf16* __restrict__ vws, const float* __restrict__ mask,
    const float* __restrict__ pos_bias, const float* __restrict__ prev,
    const int* __restrict__ flagp, float* __restrict__ prev_out,
    float* __restrict__ ctx) {
  extern __shared__ char smem[];
  float*  sc = (float*)smem;                         // 16 x 2048 fp32 scores
  __bf16* pr = (__bf16*)(smem + 16 * SS * 4);        // 16 x 2048 bf16 probs
  __shared__ float red[16][8];
  __shared__ float rowstat[16];
  __shared__ alignas(16) __bf16 vbuf[2][32 * DD];    // V double buffer (2 x 4KB)

  const int tid  = threadIdx.x;
  const int lane = tid & 31, wave = tid >> 5;
  const int q0 = blockIdx.x * 16;
  const int h  = blockIdx.y;
  const int bb = blockIdx.z;
  const int flag = *flagp;

  const __bf16* qp = qws + (((size_t)bb * HH + h) * SS + q0) * DD;
  const __bf16* kp = kws + (((size_t)bb * HH + h) * SS) * DD;
  const __bf16* vp = vws + (((size_t)bb * HH + h) * SS) * DD;

  // --- async V staging helper: chunk = 32 rows x 64 cols bf16 = 4KB, 16B/thread x2
  const unsigned vb0 = (unsigned)(size_t)&vbuf[0][0];  // LDS byte offset (low 32b)
  auto stage_v = [&](int krow, int buf) {
    const unsigned l0 = vb0 + (unsigned)buf * (32 * DD * 2) + (unsigned)tid * 16;
    const unsigned g0 = (unsigned)(krow * DD * 2) + (unsigned)tid * 16;
    asm volatile("global_load_async_to_lds_b128 %0, %1, %2"
                 :: "v"(l0), "v"(g0), "s"(vp) : "memory");
    asm volatile("global_load_async_to_lds_b128 %0, %1, %2"
                 :: "v"(l0 + 2048u), "v"(g0 + 2048u), "s"(vp) : "memory");
  };
  // Prefetch first two V chunks now; TDM-style overlap with the QK^T phase below.
  stage_v(0, 0);
  stage_v(32, 1);

  // ---- phase A: raw scores = Q K^T  -> LDS strip
  const v16bf aq0 = load_a(qp + 0, DD, lane);
  const v16bf aq1 = load_a(qp + 32, DD, lane);
  const int nlane = lane & 15, half = lane >> 4;
  for (int nt = wave; nt < SS / 16; nt += 4) {
    const int c0 = nt * 16;
    v8f c = {};
    v16bf b0 = load_bt(kp + (size_t)c0 * DD + 0, DD, lane);
    v16bf b1 = load_bt(kp + (size_t)c0 * DD + 32, DD, lane);
    c = wmma_bf16(aq0, b0, c);
    c = wmma_bf16(aq1, b1, c);
#pragma unroll
    for (int r = 0; r < 8; ++r) {
      sc[(r + 8 * half) * SS + c0 + nlane] = c[r];
    }
  }
  __syncthreads();

  // ---- phase B: fused bias/scale/mask/residual + prev_attn_out store.
  // All four global tensors are contiguous over the 16-row strip -> pure 1D
  // fully-coalesced b128 streaming (this is the HBM-dominant 1 GB of traffic).
  {
    const float* pbB = pos_bias + ((size_t)h * SS + q0) * SS;
    const float* mkB = mask + ((size_t)bb * SS + q0) * SS;
    const float* paB = prev + (((size_t)bb * HH + h) * SS + q0) * SS;
    float*       poB = prev_out + (((size_t)bb * HH + h) * SS + q0) * SS;
    for (int i = tid * 4; i < 16 * SS; i += 128 * 4) {
      fvec4 s  = *(const fvec4*)(sc + i);
      fvec4 pb = *(const fvec4*)(pbB + i);
      fvec4 mk = *(const fvec4*)(mkB + i);
      fvec4 sv = (s + pb) * 0.125f + mk;   // 1/sqrt(64)
      if (flag) sv += *(const fvec4*)(paB + i);
      *(fvec4*)(poB + i) = sv;
      *(fvec4*)(sc + i)  = sv;
    }
  }
  __syncthreads();

  // ---- phase C: softmax over each of the 16 rows (8 threads per row)
  const int row = tid >> 3, sub = tid & 7;
  float lmax = -3.4e38f;
  for (int i = sub; i < SS; i += 8) lmax = fmaxf(lmax, sc[row * SS + i]);
  red[row][sub] = lmax;
  __syncthreads();
  if (sub == 0) {
    float m = red[row][0];
#pragma unroll
    for (int j = 1; j < 8; ++j) m = fmaxf(m, red[row][j]);
    rowstat[row] = m;
  }
  __syncthreads();
  const float rmax = rowstat[row];
  float lsum = 0.f;
  for (int i = sub; i < SS; i += 8) {
    const float e = __expf(sc[row * SS + i] - rmax);
    pr[row * SS + i] = (__bf16)e;
    lsum += e;
  }
  red[row][sub] = lsum;
  __syncthreads();
  if (sub == 0) {
    float s = 0.f;
#pragma unroll
    for (int j = 0; j < 8; ++j) s += red[row][j];
    rowstat[row] = 1.0f / s;  // fold normalization into the PV accumulator
  }
  __syncthreads();

  // ---- phase D: ctx = softmax(scores) @ V, V double-buffered via async LDS copy
  const int nd0 = wave * 16;
  v8f cc = {};
  constexpr int NCH = SS / 32;  // 64 chunks
  for (int it = 0; it < NCH; ++it) {
    const int cur = it & 1;
    // In-order async completion per wave: after the next chunk is issued, at most
    // 2 ops are outstanding; waiting to <=2 proves chunk `it` has landed.
    if (it + 1 < NCH) {
      asm volatile("s_wait_asynccnt 2" ::: "memory");
    } else {
      asm volatile("s_wait_asynccnt 0" ::: "memory");
    }
    __syncthreads();  // make all waves' async LDS writes of chunk `it` visible
    v16bf a = load_a(pr + it * 32, SS, lane);             // probs from LDS
    v16bf b = load_b(&vbuf[cur][0] + nd0, DD, lane);      // V tile from LDS
    cc = wmma_bf16(a, b, cc);
    __syncthreads();  // everyone done reading buf `cur` before we overwrite it
    if (it + 2 < NCH) stage_v((it + 2) * 32, cur);
  }
#pragma unroll
  for (int r = 0; r < 8; ++r) {
    const int row2 = r + 8 * half;
    const float val = cc[r] * rowstat[row2];
    ctx[((size_t)bb * SS + q0 + row2) * HID + h * DD + nd0 + nlane] = val;
  }
}

// ---------------- kernel 3: output projection ----------------
// grid = (M/16, HID/64), block = 128. out = ctx @ Wo^T + bo
__global__ __launch_bounds__(128) void out_proj_kernel(
    const float* __restrict__ ctx, const float* __restrict__ Wo,
    const float* __restrict__ bo, float* __restrict__ out) {
  const int lane = threadIdx.x & 31, wave = threadIdx.x >> 5;
  const int m0 = blockIdx.x * 16;
  const int n0 = blockIdx.y * 64 + wave * 16;

  v8f c = {};
  for (int kk = 0; kk < HID; kk += 32) {
    v16bf a = load_a(ctx + (size_t)m0 * HID + kk, HID, lane);
    v16bf b = load_bt(Wo + (size_t)n0 * HID + kk, HID, lane);
    c = wmma_bf16(a, b, c);
  }
  const int n = lane & 15, half = lane >> 4;
  const float bn = bo[n0 + n];
#pragma unroll
  for (int r = 0; r < 8; ++r) {
    const int m = m0 + r + 8 * half;
    out[(size_t)m * HID + n0 + n] = c[r] + bn;
  }
}

// ---------------- launch ----------------
extern "C" void kernel_launch(void* const* d_in, const int* in_sizes, int n_in,
                              void* d_out, int out_size, void* d_ws,
                              size_t ws_size, hipStream_t stream) {
  (void)in_sizes; (void)n_in; (void)out_size; (void)ws_size;
  const float* key      = (const float*)d_in[0];
  const float* value    = (const float*)d_in[1];
  const float* query    = (const float*)d_in[2];
  const float* mask     = (const float*)d_in[3];
  const float* pos_bias = (const float*)d_in[4];
  const float* prev     = (const float*)d_in[5];
  const int*   flag     = (const int*)d_in[6];
  const float* Wq = (const float*)d_in[7];
  const float* bq = (const float*)d_in[8];
  const float* Wk = (const float*)d_in[9];
  const float* bk = (const float*)d_in[10];
  const float* Wv = (const float*)d_in[11];
  const float* bv = (const float*)d_in[12];
  const float* Wo = (const float*)d_in[13];
  const float* bo = (const float*)d_in[14];

  float* out      = (float*)d_out;                        // [B,S,HID]
  float* prev_out = out + (size_t)BB * SS * HID;          // [B,H,S,S]

  const size_t qkvN = (size_t)BB * HH * SS * DD;          // per-tensor elems
  __bf16* qws = (__bf16*)d_ws;
  __bf16* kws = qws + qkvN;
  __bf16* vws = kws + qkvN;
  float*  ctx = (float*)(vws + qkvN);                     // [B*S, HID] fp32

  qkv_proj_kernel<<<dim3((BB * SS) / 16, HID / 64, 3), 128, 0, stream>>>(
      query, key, value, Wq, bq, Wk, bk, Wv, bv, qws, kws, vws);

  const size_t attn_lds = (size_t)16 * SS * 4 + (size_t)16 * SS * 2;  // 192 KB
  attn_core_kernel<<<dim3(SS / 16, HH, BB), 128, attn_lds, stream>>>(
      qws, kws, vws, mask, pos_bias, prev, flag, prev_out, ctx);

  out_proj_kernel<<<dim3((BB * SS) / 16, HID / 64), 128, 0, stream>>>(
      ctx, Wo, bo, out);
}
